// GATLayer_74869869904018
// MI455X (gfx1250) — matrix operand order repaired and checked by reading, not compiled, not run on hardware
//
#include <hip/hip_runtime.h>
#include <hip/hip_bf16.h>

typedef __attribute__((ext_vector_type(16))) __bf16       v16bf;
typedef __attribute__((ext_vector_type(8)))  float        v8f;
typedef __attribute__((ext_vector_type(4)))  float        v4f;
typedef __attribute__((ext_vector_type(8)))  unsigned int v8u;

#define N_NODES 8192
#define IN_F    512
#define OUT_F   128
#define SPLITK  2

// ---- fp32 -> bf16 packing helpers ----
// RNE (for real-valued x):
__device__ __forceinline__ unsigned pkbf_rne(float a, float b) {
    unsigned ua = __float_as_uint(a), ub = __float_as_uint(b);
    ua = (ua + 0x7FFFu + ((ua >> 16) & 1u)) >> 16;
    ub = (ub + 0x7FFFu + ((ub >> 16) & 1u)) >> 16;
    return (ua & 0xFFFFu) | (ub << 16);
}
// Truncation via one v_perm_b32 (exact for {0,1}-valued adj):
// out = { b[31:16], a[31:16] }  -> sel bytes: a.b2, a.b3, b.b2, b.b3
__device__ __forceinline__ unsigned pkbf_trunc(float a, float b) {
    return __builtin_amdgcn_perm(__float_as_uint(b), __float_as_uint(a), 0x07060302u);
}
__device__ __forceinline__ unsigned short f2bf(float a) {
    unsigned u = __float_as_uint(a);
    return (unsigned short)((u + 0x7FFFu + ((u >> 16) & 1u)) >> 16);
}

// ---- core: one wave computes a 16-row x 128-col strip of (A_f32 @ B_bf16) ----
// A: row-major fp32, leading dim lda (converted to bf16 in registers)
// Bt: B transposed, row-major bf16 (ushort), leading dim ldb  (Bt[n][k] = B[k][n])
// WMMA bf16 fragment layouts used here (wave32, contiguous k, same order for A and B):
//   A 16x32: lane l<16 -> row l, K = ks+0..15 contiguous; lanes>=16 -> K = ks+16..31
//   B 32x16: lane l<16 -> col l, K = ks+0..15 contiguous; lanes>=16 -> K = ks+16..31
template <bool TRUNC, bool NTA>
__device__ __forceinline__ void gemm_strip16(const float* __restrict__ A, int lda,
                                             const unsigned short* __restrict__ Bt, int ldb,
                                             int row0, int kbeg, int kend, v8f acc[8]) {
    const int lane = threadIdx.x & 31;
    const int lh   = lane & 15;
    const int hi   = lane >> 4;
    const float* arow = A + (size_t)(row0 + lh) * (size_t)lda;

    for (int ks = kbeg; ks < kend; ks += 32) {
        const int k0 = ks + hi * 16;
        v4f f0, f1, f2, f3;
        if (NTA) {
            f0 = __builtin_nontemporal_load((const v4f*)(arow + k0));
            f1 = __builtin_nontemporal_load((const v4f*)(arow + k0 + 4));
            f2 = __builtin_nontemporal_load((const v4f*)(arow + k0 + 8));
            f3 = __builtin_nontemporal_load((const v4f*)(arow + k0 + 12));
        } else {
            f0 = *(const v4f*)(arow + k0);
            f1 = *(const v4f*)(arow + k0 + 4);
            f2 = *(const v4f*)(arow + k0 + 8);
            f3 = *(const v4f*)(arow + k0 + 12);
        }
        v8u au;
        if (TRUNC) {
            au[0] = pkbf_trunc(f0[0], f0[1]); au[1] = pkbf_trunc(f0[2], f0[3]);
            au[2] = pkbf_trunc(f1[0], f1[1]); au[3] = pkbf_trunc(f1[2], f1[3]);
            au[4] = pkbf_trunc(f2[0], f2[1]); au[5] = pkbf_trunc(f2[2], f2[3]);
            au[6] = pkbf_trunc(f3[0], f3[1]); au[7] = pkbf_trunc(f3[2], f3[3]);
        } else {
            au[0] = pkbf_rne(f0[0], f0[1]); au[1] = pkbf_rne(f0[2], f0[3]);
            au[2] = pkbf_rne(f1[0], f1[1]); au[3] = pkbf_rne(f1[2], f1[3]);
            au[4] = pkbf_rne(f2[0], f2[1]); au[5] = pkbf_rne(f2[2], f2[3]);
            au[6] = pkbf_rne(f3[0], f3[1]); au[7] = pkbf_rne(f3[2], f3[3]);
        }
        v16bf av = __builtin_bit_cast(v16bf, au);

#pragma unroll
        for (int t = 0; t < 8; ++t) {
            const v8u* bp = (const v8u*)(Bt + (size_t)(16 * t + lh) * (size_t)ldb + ks + hi * 16);
            v16bf bv = __builtin_bit_cast(v16bf, *bp);
            acc[t] = __builtin_amdgcn_wmma_f32_16x16x32_bf16(
                false, av, false, bv, (short)0, acc[t], false, false);
        }
    }
}

// ---- kernel 0: Wt[f][k] = bf16(W[k][f]) ----
__global__ void wt_kernel(const float* __restrict__ W, unsigned short* __restrict__ Wt) {
    int idx = blockIdx.x * 256 + threadIdx.x;      // 128*512 = 65536 threads
    int f = idx >> 9, k = idx & 511;
    Wt[idx] = f2bf(W[(size_t)k * OUT_F + f]);
}

// ---- kernel 1: h = x @ W  (bf16 WMMA, f32 accum). 4 waves/WG, 16 rows/wave ----
__global__ void fc_kernel(const float* __restrict__ x, const unsigned short* __restrict__ Wt,
                          float* __restrict__ h) {
    const int wave = threadIdx.x >> 5;
    const int lane = threadIdx.x & 31, lh = lane & 15, hi = lane >> 4;
    const int row0 = blockIdx.x * 64 + wave * 16;
    v8f acc[8];
#pragma unroll
    for (int t = 0; t < 8; ++t)
#pragma unroll
        for (int v = 0; v < 8; ++v) acc[t][v] = 0.0f;

    gemm_strip16<false, false>(x, IN_F, Wt, IN_F, row0, 0, IN_F, acc);

#pragma unroll
    for (int t = 0; t < 8; ++t)
#pragma unroll
        for (int v = 0; v < 8; ++v)
            h[(size_t)(row0 + 8 * hi + v) * OUT_F + 16 * t + lh] = acc[t][v];
}

// ---- kernel 2: s[j] = h[j,:] . a[F:2F]  (one wave per row) ----
__global__ void ssrc_kernel(const float* __restrict__ h, const float* __restrict__ a,
                            float* __restrict__ s) {
    const int j    = blockIdx.x * 8 + (threadIdx.x >> 5);
    const int lane = threadIdx.x & 31;
    v4f hv = *(const v4f*)(h + (size_t)j * OUT_F + lane * 4);
    v4f av = *(const v4f*)(a + OUT_F + lane * 4);
    float d = hv[0] * av[0] + hv[1] * av[1] + hv[2] * av[2] + hv[3] * av[3];
#pragma unroll
    for (int o = 16; o > 0; o >>= 1) d += __shfl_xor(d, o, 32);
    if (lane == 0) s[j] = d;
}

// ---- kernel 3: p = softmax(s) over all 8192 entries (single block) ----
__global__ void softmax_kernel(const float* __restrict__ s, float* __restrict__ p) {
    __shared__ float red[1024];
    const int tid = threadIdx.x;
    float m = -3.4e38f;
    for (int i = tid; i < N_NODES; i += 1024) m = fmaxf(m, s[i]);
    red[tid] = m; __syncthreads();
    for (int o = 512; o > 0; o >>= 1) {
        if (tid < o) red[tid] = fmaxf(red[tid], red[tid + o]);
        __syncthreads();
    }
    m = red[0]; __syncthreads();
    float sum = 0.0f;
    for (int i = tid; i < N_NODES; i += 1024) {
        float e = expf(s[i] - m);
        p[i] = e;
        sum += e;
    }
    red[tid] = sum; __syncthreads();
    for (int o = 512; o > 0; o >>= 1) {
        if (tid < o) red[tid] += red[tid + o];
        __syncthreads();
    }
    float inv = 1.0f / red[0];
    for (int i = tid; i < N_NODES; i += 1024) p[i] *= inv;
}

// ---- kernel 4: gT[f][j] = bf16(p[j] * h[j][f])  (B-operand, transposed) ----
__global__ void scale_t_kernel(const float* __restrict__ h, const float* __restrict__ p,
                               unsigned short* __restrict__ gT) {
    int idx = blockIdx.x * 256 + threadIdx.x;      // 128*8192 = 1M threads
    int f = idx >> 13, j = idx & (N_NODES - 1);
    gT[idx] = f2bf(p[j] * h[(size_t)j * OUT_F + f]);
}

// ---- kernel 5: part[split] = adj[:, kseg] @ g[kseg, :]  (main WMMA GEMM) ----
__global__ void aggr_kernel(const float* __restrict__ adj, const unsigned short* __restrict__ gT,
                            float* __restrict__ part) {
    const int wave  = threadIdx.x >> 5;
    const int lane  = threadIdx.x & 31, lh = lane & 15, hi = lane >> 4;
    const int rwg   = blockIdx.x & 127;            // 128 row-workgroups (64 rows each)
    const int split = blockIdx.x >> 7;
    const int row0  = rwg * 64 + wave * 16;
    const int kbeg  = split * (N_NODES / SPLITK);
    const int kend  = kbeg + (N_NODES / SPLITK);

    v8f acc[8];
#pragma unroll
    for (int t = 0; t < 8; ++t)
#pragma unroll
        for (int v = 0; v < 8; ++v) acc[t][v] = 0.0f;

    // adj is streamed once -> non-temporal; adj entries are {0,1} -> truncation exact
    gemm_strip16<true, true>(adj, N_NODES, gT, N_NODES, row0, kbeg, kend, acc);

    float* outp = part + (size_t)split * N_NODES * OUT_F;
#pragma unroll
    for (int t = 0; t < 8; ++t)
#pragma unroll
        for (int v = 0; v < 8; ++v)
            outp[(size_t)(row0 + 8 * hi + v) * OUT_F + 16 * t + lh] = acc[t][v];
}

// ---- kernel 6: out = elu(sum of split partials) ----
__global__ void elu_kernel(const float* __restrict__ part, float* __restrict__ out) {
    int idx = blockIdx.x * 256 + threadIdx.x;      // 1M threads
    float v = 0.0f;
#pragma unroll
    for (int sp = 0; sp < SPLITK; ++sp) v += part[(size_t)sp * N_NODES * OUT_F + idx];
    out[idx] = (v > 0.0f) ? v : expm1f(v);
}

extern "C" void kernel_launch(void* const* d_in, const int* in_sizes, int n_in,
                              void* d_out, int out_size, void* d_ws, size_t ws_size,
                              hipStream_t stream) {
    const float* x   = (const float*)d_in[0];   // [8192, 512]
    const float* adj = (const float*)d_in[1];   // [8192, 8192]
    const float* W   = (const float*)d_in[2];   // [512, 128]
    const float* a   = (const float*)d_in[3];   // [256]

    char* ws = (char*)d_ws;
    float*          h    = (float*)(ws);                                        // 4 MB
    unsigned short* Wt   = (unsigned short*)(ws + (4u << 20));                  // 128 KB
    float*          s    = (float*)(ws + (4u << 20) + (128u << 10));            // 32 KB
    float*          p    = (float*)(ws + (4u << 20) + (160u << 10));            // 32 KB
    unsigned short* gT   = (unsigned short*)(ws + (4u << 20) + (192u << 10));   // 2 MB
    float*          part = (float*)(ws + (6u << 20) + (192u << 10));            // SPLITK * 4 MB
    float*          out  = (float*)d_out;

    wt_kernel     <<<256,            256, 0, stream>>>(W, Wt);
    fc_kernel     <<<128,            128, 0, stream>>>(x, Wt, h);
    ssrc_kernel   <<<1024,           256, 0, stream>>>(h, a, s);
    softmax_kernel<<<1,             1024, 0, stream>>>(s, p);
    scale_t_kernel<<<4096,           256, 0, stream>>>(h, p, gT);
    aggr_kernel   <<<128 * SPLITK,   128, 0, stream>>>(adj, gT, part);
    elu_kernel    <<<4096,           256, 0, stream>>>(part, out);
}
